// MultiheadAttentionWithRelativePosition_53712861004260
// MI455X (gfx1250) — matrix operand order repaired and checked by reading, not compiled, not run on hardware
//
#include <hip/hip_runtime.h>
#include <stdint.h>

#define BB 8
#define LL 512
#define DD 512
#define HH 8
#define DK 64
#define NREL 129      // 2*64+1 relative buckets
#define RK_N 144      // Qrel N padded to 9*16
#define RV_K 160      // attnR K padded to 5*32

typedef __bf16 bf16;
typedef bf16  v16bf __attribute__((ext_vector_type(16)));
typedef float v8f   __attribute__((ext_vector_type(8)));

union Frag { v16bf v; unsigned int u[8]; };

__device__ __forceinline__ unsigned short f2bf(float f) {
  unsigned int u = __float_as_uint(f);
  u += 0x7fffu + ((u >> 16) & 1u);            // round-to-nearest-even
  return (unsigned short)(u >> 16);
}

__device__ __forceinline__ v8f vzero() {
  v8f z;
#pragma unroll
  for (int i = 0; i < 8; ++i) z[i] = 0.0f;
  return z;
}

__device__ __forceinline__ v8f wmma_bf16(const Frag& a, const Frag& b, v8f c) {
  return __builtin_amdgcn_wmma_f32_16x16x32_bf16(
      /*neg_a=*/false, a.v, /*neg_b=*/false, b.v,
      /*c_mod=*/(short)0, c, /*reuse_a=*/false, /*reuse_b=*/false);
}

// ---------------- pre-pass kernels ----------------

__global__ void k_cvt_bf16(const float* __restrict__ src,
                           unsigned short* __restrict__ dst, int n) {
  int i = blockIdx.x * blockDim.x + threadIdx.x;
  if (i < n) dst[i] = f2bf(src[i]);
}

// Pack W[K,N] f32 -> pair-packed bf16: WP[t][n] = {W[2t+1][n], W[2t][n]}
__global__ void k_pack_w(const float* __restrict__ W,
                         unsigned int* __restrict__ WP, int Kdim, int N) {
  int i = blockIdx.x * blockDim.x + threadIdx.x;
  int total = (Kdim >> 1) * N;
  if (i >= total) return;
  int t = i / N, n = i % N;
  unsigned int lo = f2bf(W[(2 * t) * N + n]);
  unsigned int hi = f2bf(W[(2 * t + 1) * N + n]);
  WP[i] = lo | (hi << 16);
}

// rel_k[r][h][d] -> relkP[h][t][r] = {rel_k[r][h][2t+1], rel_k[r][h][2t]}, r padded to 144
__global__ void k_pack_relk(const float* __restrict__ rel_k,
                            unsigned int* __restrict__ relkP) {
  int i = blockIdx.x * blockDim.x + threadIdx.x;
  int total = HH * 32 * RK_N;
  if (i >= total) return;
  int h = i / (32 * RK_N);
  int rem = i % (32 * RK_N);
  int t = rem / RK_N, r = rem % RK_N;
  unsigned int lo = 0, hi = 0;
  if (r < NREL) {
    lo = f2bf(rel_k[((size_t)r * HH + h) * DK + 2 * t]);
    hi = f2bf(rel_k[((size_t)r * HH + h) * DK + 2 * t + 1]);
  }
  relkP[i] = lo | (hi << 16);
}

// rel_v[r][h][d] -> relvP[h][t][d] = {rel_v[2t+1][h][d], rel_v[2t][h][d]}, K padded to 160
__global__ void k_pack_relv(const float* __restrict__ rel_v,
                            unsigned int* __restrict__ relvP) {
  int i = blockIdx.x * blockDim.x + threadIdx.x;
  int total = HH * (RV_K / 2) * DK;
  if (i >= total) return;
  int h = i / ((RV_K / 2) * DK);
  int rem = i % ((RV_K / 2) * DK);
  int t = rem / DK, d = rem % DK;
  int r0 = 2 * t, r1 = 2 * t + 1;
  unsigned int lo = (r0 < NREL) ? f2bf(rel_v[((size_t)r0 * HH + h) * DK + d]) : 0u;
  unsigned int hi = (r1 < NREL) ? f2bf(rel_v[((size_t)r1 * HH + h) * DK + d]) : 0u;
  relvP[i] = lo | (hi << 16);
}

// ---------------- generic projection GEMM ----------------
// C[M=4096, N=512] = A(bf16 row-major) x W(pair-packed bf16) + bias
// One block = one 16-row tile; A tile staged to LDS via async-to-LDS DMA,
// 8 waves = 8 column groups of 64 share it. B loads are lane-consecutive dwords.
// mode 0: scatter bf16 into [B,H,L,DK] head layout (for Q/K)
// mode 1: write f32 row-major (final output @ Wo)
__global__ void __launch_bounds__(256)
k_proj_gemm(const unsigned short* __restrict__ A,
            const unsigned int* __restrict__ WP,   // [256][512]
            const float* __restrict__ bias,
            unsigned short* __restrict__ outHeads,
            float* __restrict__ outF, int mode) {
  __shared__ __align__(16) unsigned short Atile[16 * DD];  // 16 KB

  const int tid = threadIdx.x;
  const int lane = tid & 31;
  const int wave = tid >> 5;          // 0..7 -> column group
  const int rowTile = blockIdx.x;     // 0..255
  const int nl = lane & 15;
  const int msel = lane >> 4;

  // --- stage A tile (16 x 512 bf16 = 16384 B) into LDS, async DMA path ---
  {
    const char* gbase = (const char*)A + (size_t)rowTile * 16 * DD * 2;
#pragma unroll
    for (int i = 0; i < 4; ++i) {
      int off = tid * 16 + i * 4096;
      unsigned ldsOff = (unsigned)(uintptr_t)((char*)Atile + off);
      unsigned long long ga = (unsigned long long)(uintptr_t)(gbase + off);
      asm volatile("global_load_async_to_lds_b128 %0, %1, off"
                   :: "v"(ldsOff), "v"(ga) : "memory");
    }
    asm volatile("s_wait_asynccnt 0" ::: "memory");
    __syncthreads();
  }

  const int cg = wave;
  v8f acc[4];
#pragma unroll
  for (int t = 0; t < 4; ++t) acc[t] = vzero();

  for (int ks = 0; ks < 16; ++ks) {                      // K=512, 32/step
    Frag a;
#pragma unroll
    for (int j = 0; j < 8; ++j) {
      int k = ks * 32 + (msel ? 16 : 0) + 2 * j;
      a.u[j] = *(const unsigned int*)(&Atile[nl * DD + k]);   // ds_load_b32
    }
#pragma unroll
    for (int nt = 0; nt < 4; ++nt) {
      Frag bfr;
      int n = cg * 64 + nt * 16 + nl;
#pragma unroll
      for (int j = 0; j < 8; ++j) {
        int t = ks * 16 + (msel ? 8 : 0) + j;            // K-row pair index
        bfr.u[j] = WP[t * DD + n];
      }
      acc[nt] = wmma_bf16(a, bfr, acc[nt]);
    }
  }

#pragma unroll
  for (int nt = 0; nt < 4; ++nt) {
    int n = cg * 64 + nt * 16 + nl;
    float bv = bias[n];
#pragma unroll
    for (int r = 0; r < 8; ++r) {
      int mr = rowTile * 16 + r + msel * 8;
      float val = acc[nt][r] + bv;
      if (mode == 0) {
        int h = n >> 6, d = n & 63;
        int bidx = mr >> 9, l = mr & 511;
        outHeads[(((size_t)(bidx * HH + h)) * LL + l) * DK + d] = f2bf(val);
      } else {
        outF[(size_t)mr * DD + n] = val;
      }
    }
  }
}

// ---------------- fused relative attention ----------------
// One block = one (b, h, 16-row tile): scores via WMMA, Qrel via WMMA,
// softmax fp32, bucket-sum to attnR, attnR @ rel_v via WMMA.
__global__ void __launch_bounds__(128)
k_attn(const unsigned short* __restrict__ Qb,   // [B,H,L,DK] bf16
       const unsigned short* __restrict__ Kb,   // [B,H,L,DK] bf16
       const unsigned int* __restrict__ relkP,  // [H][32][144]
       const unsigned int* __restrict__ relvP,  // [H][80][64]
       unsigned short* __restrict__ outHb) {    // [B,L,512] bf16
  __shared__ float S[16][LL];                   // 32 KB scores/attn
  __shared__ float Qr[16][RK_N];                // 9 KB Qrel
  __shared__ unsigned int aR[16][RV_K / 2];     // 5 KB attnR (bf16 pairs)
  __shared__ float red[16][16];                 // reductions

  const int lane = threadIdx.x & 31;
  const int wave = threadIdx.x >> 5;
  const int nl = lane & 15;
  const int msel = lane >> 4;

  const int ltile = blockIdx.x & 31;
  const int bh = blockIdx.x >> 5;
  const int h = bh & 7;
  const int b = bh >> 3;
  const int l0 = ltile * 16;
  const size_t headBase = ((size_t)(b * HH + h)) * LL * DK;

  // A-fragments of Q rows l0..l0+15 (shared by content + Qrel GEMMs)
  Frag aq[2];
#pragma unroll
  for (int ks = 0; ks < 2; ++ks)
#pragma unroll
    for (int j = 0; j < 8; ++j) {
      int k = ks * 32 + (msel ? 16 : 0) + 2 * j;
      aq[ks].u[j] = *(const unsigned int*)(Qb + headBase + (size_t)(l0 + nl) * DK + k);
    }

  // Content scores S = Q K^T : B(k,n)=K[m0+n,k] -> contiguous dwords of row-major K
  for (int mt = wave; mt < 32; mt += 4) {
    v8f acc = vzero();
#pragma unroll
    for (int ks = 0; ks < 2; ++ks) {
      Frag bfr;
#pragma unroll
      for (int j = 0; j < 8; ++j) {
        int k = ks * 32 + (msel ? 16 : 0) + 2 * j;
        bfr.u[j] = *(const unsigned int*)(Kb + headBase + (size_t)(mt * 16 + nl) * DK + k);
      }
      acc = wmma_bf16(aq[ks], bfr, acc);
    }
#pragma unroll
    for (int r = 0; r < 8; ++r) S[r + msel * 8][mt * 16 + nl] = acc[r];
  }

  // Qrel[l, r] = Q . rel_k[r,h,:]   (N = 144 padded, K = 64)
  for (int nt = wave; nt < 9; nt += 4) {
    v8f acc = vzero();
#pragma unroll
    for (int ks = 0; ks < 2; ++ks) {
      Frag bfr;
#pragma unroll
      for (int j = 0; j < 8; ++j) {
        int t = ks * 16 + (msel ? 8 : 0) + j;
        bfr.u[j] = relkP[(h * 32 + t) * RK_N + nt * 16 + nl];
      }
      acc = wmma_bf16(aq[ks], bfr, acc);
    }
#pragma unroll
    for (int r = 0; r < 8; ++r) Qr[r + msel * 8][nt * 16 + nl] = acc[r];
  }
  __syncthreads();

  // Softmax over m (8 threads per row), with relative-score lookup fused in
  const int lrow = threadIdx.x >> 3;
  const int s8 = threadIdx.x & 7;
  const int labs = l0 + lrow;
  float mx = -3.0e38f;
  for (int mc = s8; mc < LL; mc += 8) {
    int ri = labs - mc;
    ri = ri < -64 ? -64 : (ri > 64 ? 64 : ri);
    float v = (S[lrow][mc] + Qr[lrow][ri + 64]) * 0.125f;  // 1/sqrt(64)
    S[lrow][mc] = v;
    mx = fmaxf(mx, v);
  }
  red[lrow][s8] = mx;
  __syncthreads();
  float mAll = red[lrow][0];
#pragma unroll
  for (int i = 1; i < 8; ++i) mAll = fmaxf(mAll, red[lrow][i]);
  float sum = 0.0f;
  for (int mc = s8; mc < LL; mc += 8) {
    float e = __expf(S[lrow][mc] - mAll);
    S[lrow][mc] = e;
    sum += e;
  }
  red[lrow][8 + s8] = sum;
  __syncthreads();
  float tot = 0.0f;
#pragma unroll
  for (int i = 0; i < 8; ++i) tot += red[lrow][8 + i];
  float inv = 1.0f / tot;

  // Bucket-sum: attnR[l,r] ; interior r <-> m = labs+64-r, edges are range sums
  unsigned short* aRh = (unsigned short*)aR;   // row stride RV_K u16
  for (int r = 1 + s8; r < 128; r += 8) {
    int mc = labs + 64 - r;
    float v = (mc >= 0 && mc < LL) ? S[lrow][mc] * inv : 0.0f;
    aRh[lrow * RV_K + r] = f2bf(v);
  }
  if (s8 == 0) {
    float a0 = 0.0f;
    for (int mc = labs + 64; mc < LL; ++mc) a0 += S[lrow][mc];
    aRh[lrow * RV_K + 0] = f2bf(a0 * inv);
  }
  if (s8 == 1) {
    float a1 = 0.0f;
    for (int mc = 0; mc <= labs - 64; ++mc) a1 += S[lrow][mc];
    aRh[lrow * RV_K + 128] = f2bf(a1 * inv);
  }
  for (int r = 129 + s8; r < RV_K; r += 8) aRh[lrow * RV_K + r] = 0;
  __syncthreads();

  // out[16,64] = attnR[16,160] @ rel_v ; one 16-col tile per wave
  {
    const int nt = wave;
    v8f acc = vzero();
    for (int ks = 0; ks < 5; ++ks) {
      Frag a, bfr;
#pragma unroll
      for (int j = 0; j < 8; ++j) {
        int kp = ks * 16 + (msel ? 8 : 0) + j;
        a.u[j] = aR[nl][kp];
      }
#pragma unroll
      for (int j = 0; j < 8; ++j) {
        int t = ks * 16 + (msel ? 8 : 0) + j;
        bfr.u[j] = relvP[(h * (RV_K / 2) + t) * DK + nt * 16 + nl];
      }
      acc = wmma_bf16(a, bfr, acc);
    }
    int d = nt * 16 + nl;
#pragma unroll
    for (int r = 0; r < 8; ++r) {
      int lr = l0 + r + msel * 8;
      outHb[((size_t)(b * LL + lr)) * DD + h * DK + d] = f2bf(acc[r]);
    }
  }
}

// ---------------- launcher ----------------
extern "C" void kernel_launch(void* const* d_in, const int* in_sizes, int n_in,
                              void* d_out, int out_size, void* d_ws, size_t ws_size,
                              hipStream_t stream) {
  const float* x    = (const float*)d_in[0];
  const float* Wq   = (const float*)d_in[1];
  const float* bq   = (const float*)d_in[2];
  const float* Wk   = (const float*)d_in[3];
  const float* bk   = (const float*)d_in[4];
  // d_in[5]=Wv, d_in[6]=bv: unused — reference discards the attn@V term.
  const float* Wo   = (const float*)d_in[7];
  const float* bo   = (const float*)d_in[8];
  const float* relk = (const float*)d_in[9];
  const float* relv = (const float*)d_in[10];
  float* out = (float*)d_out;

  char* ws = (char*)d_ws;
  unsigned short* xb  = (unsigned short*)(ws + 0);         // 4 MB
  unsigned short* Qb  = (unsigned short*)(ws + 4194304);   // 4 MB
  unsigned short* Kb  = (unsigned short*)(ws + 8388608);   // 4 MB
  unsigned short* oHb = (unsigned short*)(ws + 12582912);  // 4 MB
  unsigned int* WqP   = (unsigned int*)(ws + 16777216);    // 512 KB
  unsigned int* WkP   = (unsigned int*)(ws + 17301504);
  unsigned int* WoP   = (unsigned int*)(ws + 17825792);
  unsigned int* relkP = (unsigned int*)(ws + 18350080);    // 144 KB
  unsigned int* relvP = (unsigned int*)(ws + 18497536);    // 160 KB

  const int nx = BB * LL * DD;
  k_cvt_bf16<<<(nx + 255) / 256, 256, 0, stream>>>(x, xb, nx);

  const int nw = 256 * 512;
  k_pack_w<<<(nw + 255) / 256, 256, 0, stream>>>(Wq, WqP, 512, 512);
  k_pack_w<<<(nw + 255) / 256, 256, 0, stream>>>(Wk, WkP, 512, 512);
  k_pack_w<<<(nw + 255) / 256, 256, 0, stream>>>(Wo, WoP, 512, 512);

  const int nrk = HH * 32 * RK_N;
  k_pack_relk<<<(nrk + 255) / 256, 256, 0, stream>>>(relk, relkP);
  const int nrv = HH * (RV_K / 2) * DK;
  k_pack_relv<<<(nrv + 255) / 256, 256, 0, stream>>>(relv, relvP);

  // Q and K projections (WMMA GEMM, LDS-staged A, head-scatter bf16 output)
  k_proj_gemm<<<256, 256, 0, stream>>>(xb, WqP, bq, Qb, nullptr, 0);
  k_proj_gemm<<<256, 256, 0, stream>>>(xb, WkP, bk, Kb, nullptr, 0);

  // Fused relative attention: 8*8*32 = 2048 tiles
  k_attn<<<BB * HH * (LL / 16), 128, 0, stream>>>(Qb, Kb, relkP, relvP, oHb);

  // Final projection @ Wo + bo -> f32 output
  k_proj_gemm<<<256, 256, 0, stream>>>(oHb, WoP, bo, nullptr, out, 1);
}
